// RPN_49271864820064
// MI455X (gfx1250) — matrix operand order repaired and checked by reading, not compiled, not run on hardware
//
#include <hip/hip_runtime.h>
#include <hip/hip_bf16.h>

typedef __attribute__((ext_vector_type(16))) _Float16 v16h;
typedef __attribute__((ext_vector_type(8)))  float    v8f;
typedef __attribute__((ext_vector_type(4)))  unsigned int v4u;

#define NPOS   4096      // 64*64 positions per batch
#define NANCH  9
#define NPROP  36864     // NPOS * NANCH
#define PRE_NMS 6000
#define POST_NMS 300

// ---------------- workspace layout (bytes) ----------------
#define OFF_FEATH  ((size_t)0)                                  // f16 [8][512][64][64]
#define OFF_WP     (OFF_FEATH + (size_t)8*512*4096*2)           // f16 conv3x3 fragments
#define OFF_RPN    (OFF_WP    + (size_t)32*9*16*32*16*2)        // f16 NHWC [8][64][64][512]
#define OFF_WP1    (OFF_RPN   + (size_t)8*4096*512*2)           // f16 1x1 fragments
#define OFF_OUT1   (OFF_WP1   + (size_t)16*4*32*16*2)           // f32 [8][4096][64]
#define OFF_PROB   (OFF_OUT1  + (size_t)8*4096*64*4)            // f32 [8][36864]
#define OFF_BOX    (OFF_PROB  + (size_t)8*36864*4)              // f32 [8][36864][4]
#define OFF_CAND   (OFF_BOX   + (size_t)8*36864*4*4)            // f32 [8][6][6000]

// ---------------- precision conversion ----------------
__global__ void cvt_feat_kernel(const float* __restrict__ x, _Float16* __restrict__ y, int n) {
    int i = blockIdx.x * 256 + threadIdx.x;
    if (i < n) y[i] = (_Float16)x[i];
}

// Pack conv_w[co][ci][3][3] into WMMA B fragments:
// wp[cob(32)][tap(9)][kc(16)][lane(32)][h(16)]
// lane: N = lane&15 (out-channel), K = (h&8)*2 + (lane>>4)*8 + (h&7) (in-channel within chunk)
__global__ void pack_convw_kernel(const float* __restrict__ w, _Float16* __restrict__ wp) {
    int i = blockIdx.x * 256 + threadIdx.x;
    if (i >= 32*9*16*32*16) return;
    int hh = i & 15;
    int t = i >> 4;
    int lane = t & 31; t >>= 5;
    int kc = t & 15;  t >>= 4;
    int tap = t % 9;
    int cob = t / 9;
    int co = cob*16 + (lane & 15);
    int ci = kc*32 + (hh >> 3)*16 + (lane >> 4)*8 + (hh & 7);
    int kh = tap / 3, kw = tap % 3;
    wp[i] = (_Float16)w[(((size_t)co*512 + ci)*3 + kh)*3 + kw];
}

// Pack [cls_w(18,512) ; bbox_w(36,512) ; zero pad to 64] into wp1[kc(16)][nt(4)][lane(32)][16]
__global__ void pack_w1_kernel(const float* __restrict__ clsw, const float* __restrict__ bboxw,
                               _Float16* __restrict__ wp1) {
    int i = blockIdx.x * 256 + threadIdx.x;
    if (i >= 16*4*32*16) return;
    int hh = i & 15;
    int t = i >> 4;
    int lane = t & 31; t >>= 5;
    int nt = t & 3;
    int kc = t >> 2;
    int n  = nt*16 + (lane & 15);
    int ci = kc*32 + (hh >> 3)*16 + (lane >> 4)*8 + (hh & 7);
    float v = 0.f;
    if (n < 18)       v = clsw[(size_t)n*512 + ci];
    else if (n < 54)  v = bboxw[(size_t)(n-18)*512 + ci];
    wp1[i] = (_Float16)v;
}

// ---------------- 3x3 conv 512->512 + ReLU, implicit GEMM over WMMA ----------------
// grid (4 co-blocks of 128, 64 rows, 8 batches), 256 threads = 8 waves.
// wave: M-tile = 16 w-positions (mt), N = 4 tiles of 16 co (ng*4..ng*4+3).
__global__ __launch_bounds__(256) void conv3x3_kernel(
    const _Float16* __restrict__ feat,   // [B][512][64][64] f16
    const _Float16* __restrict__ wp,     // packed fragments
    const float*    __restrict__ convb,  // [512]
    _Float16*       __restrict__ rpn)    // [B][64][64][512] NHWC f16
{
    const int b  = blockIdx.z;
    const int h  = blockIdx.y;
    const int cb = blockIdx.x;               // co block: 128 channels
    const int tid  = threadIdx.x;
    const int lane = tid & 31;
    const int wid  = tid >> 5;
    const int mt = wid & 3;                  // w-range mt*16 .. mt*16+15
    const int ng = wid >> 2;                 // N group: tiles ng*4 .. ng*4+3
    const int m     = lane & 15;
    const int khalf = lane >> 4;

    __shared__ __align__(16) _Float16 Xs[3*66*32];   // [row][w+1][ci]

    v8f acc[4] = {};

    for (int kc = 0; kc < 16; ++kc) {
        __syncthreads();
        // stage 3 rows x 66 halo-cols x 32 in-channels, K-innermost
        for (int idx = tid; idx < 3*66*32; idx += 256) {
            int ci   = idx & 31;
            int rest = idx >> 5;
            int w    = rest % 66;
            int row  = rest / 66;
            int gh = h - 1 + row;
            int gw = w - 1;
            _Float16 v = (_Float16)0.f;
            if ((unsigned)gh < 64u && (unsigned)gw < 64u)
                v = feat[(((size_t)b*512 + (kc*32 + ci))*64 + gh)*64 + gw];
            Xs[idx] = v;
        }
        __syncthreads();

        if (kc + 1 < 16)  // warm L2/L0 for next chunk's weight stream
            __builtin_prefetch(&wp[((((size_t)(cb*8 + ng*4)*9)*16 + (kc+1))*32 + lane)*16], 0, 1);

        #pragma unroll
        for (int tap = 0; tap < 9; ++tap) {
            const int dh = tap / 3, dw = tap % 3;
            const int base = (dh*66 + (mt*16 + m + dw))*32 + khalf*8;
            union { v16h h; v4u u[2]; } af;
            af.u[0] = *(const v4u*)&Xs[base];        // K  0-7  / 8-15
            af.u[1] = *(const v4u*)&Xs[base + 16];   // K 16-23 / 24-31
            #pragma unroll
            for (int j = 0; j < 4; ++j) {
                const int nt = ng*4 + j;
                const v16h bf = *(const v16h*)&wp[((((size_t)(cb*8 + nt)*9 + tap)*16 + kc)*32 + lane)*16];
                acc[j] = __builtin_amdgcn_wmma_f32_16x16x32_f16(
                    false, af.h, false, bf, (short)0, acc[j], false, false);
            }
        }
    }

    // epilogue: bias + ReLU, store NHWC f16
    #pragma unroll
    for (int j = 0; j < 4; ++j) {
        const int co = cb*128 + (ng*4 + j)*16 + (lane & 15);
        const float bias = convb[co];
        #pragma unroll
        for (int r = 0; r < 8; ++r) {
            const int wo = mt*16 + r + 8*(lane >> 4);
            float v = acc[j][r] + bias;
            v = v > 0.f ? v : 0.f;
            rpn[(((size_t)b*64 + h)*64 + wo)*512 + co] = (_Float16)v;
        }
    }
}

// ---------------- fused 1x1 convs (cls 18 + bbox 36, N padded to 64) ----------------
// grid (32, 8 batches), 256 threads = 8 waves; wave = one 16-position M-tile.
__global__ __launch_bounds__(256) void conv1x1_kernel(
    const _Float16* __restrict__ rpn,    // [B][4096][512] NHWC
    const _Float16* __restrict__ wp1,    // packed fragments
    const float* __restrict__ clsb, const float* __restrict__ bboxb,
    float* __restrict__ out1)            // [B][4096][64]
{
    const int b    = blockIdx.y;
    const int lane = threadIdx.x & 31;
    const int wid  = threadIdx.x >> 5;
    const int pt   = blockIdx.x * 8 + wid;   // 0..255 position tiles
    const int m     = lane & 15;
    const int khalf = lane >> 4;

    v8f acc[4] = {};
    const size_t rowbase = ((size_t)b*NPOS + pt*16 + m) * 512;

    for (int kc = 0; kc < 16; ++kc) {
        union { v16h h; v4u u[2]; } af;
        const _Float16* ap = rpn + rowbase + kc*32 + khalf*8;
        af.u[0] = *(const v4u*)ap;
        af.u[1] = *(const v4u*)(ap + 16);
        #pragma unroll
        for (int j = 0; j < 4; ++j) {
            const v16h bf = *(const v16h*)&wp1[(((size_t)kc*4 + j)*32 + lane)*16];
            acc[j] = __builtin_amdgcn_wmma_f32_16x16x32_f16(
                false, af.h, false, bf, (short)0, acc[j], false, false);
        }
    }

    #pragma unroll
    for (int j = 0; j < 4; ++j) {
        const int c = j*16 + (lane & 15);
        float bias = 0.f;
        bool valid = true;
        if (c < 18)      bias = clsb[c];
        else if (c < 54) bias = bboxb[c - 18];
        else             valid = false;
        if (!valid) continue;
        #pragma unroll
        for (int r = 0; r < 8; ++r) {
            const int pos = pt*16 + r + 8*(lane >> 4);
            out1[((size_t)b*NPOS + pos)*64 + c] = acc[j][r] + bias;
        }
    }
}

// ---------------- softmax + anchor decode + clip ----------------
__global__ void decode_kernel(const float* __restrict__ out1,
                              float* __restrict__ score_out,  // [8][36864][2]
                              float* __restrict__ tgt_out,    // [8][36864][4]
                              float* __restrict__ prob,       // [8][36864]
                              float* __restrict__ boxes)      // [8][36864][4]
{
    int i = blockIdx.x * 256 + threadIdx.x;
    if (i >= 8 * NPROP) return;
    const int b  = i / NPROP;
    const int p  = i % NPROP;
    const int hw = p / NANCH;
    const int a  = p % NANCH;
    const float* row = out1 + ((size_t)b*NPOS + hw)*64;

    const float s0 = row[2*a], s1 = row[2*a + 1];
    score_out[(size_t)i*2 + 0] = s0;
    score_out[(size_t)i*2 + 1] = s1;
    const float t0 = row[18 + 4*a], t1 = row[19 + 4*a];
    const float t2 = row[20 + 4*a], t3 = row[21 + 4*a];
    tgt_out[(size_t)i*4 + 0] = t0;
    tgt_out[(size_t)i*4 + 1] = t1;
    tgt_out[(size_t)i*4 + 2] = t2;
    tgt_out[(size_t)i*4 + 3] = t3;

    prob[i] = 1.f / (1.f + expf(s0 - s1));   // softmax(...)[1]

    // regenerate ANCHORS[p] analytically: p = d*4096 + xi*64 + yj
    const int d   = p >> 12;
    const int rem = p & 4095;
    const int xi  = rem >> 6;
    const int yj  = rem & 63;
    const float scl[3] = {8.f, 16.f, 32.f};
    const float sqr[3] = {0.70710678118f, 1.f, 1.41421356237f}; // sqrt(ratio)
    const float S  = scl[d / 3];
    const float sr = sqr[d % 3];
    const float ah = 16.f * S * sr;      // dims[0]
    const float aw = 16.f * S / sr;      // dims[1]
    const float acy = 8.f + 16.f * xi;   // ctrs[:,0]
    const float acx = 8.f + 16.f * yj;   // ctrs[:,1]

    const float pcy = acy + t0 * ah;
    const float pcx = acx + t1 * aw;
    const float ph  = ah * expf(t2);
    const float pw  = aw * expf(t3);
    float y1 = pcy - 0.5f*ph, x1 = pcx - 0.5f*pw;
    float y2 = pcy + 0.5f*ph, x2 = pcx + 0.5f*pw;
    y1 = fminf(fmaxf(y1, 0.f), 1023.f);
    x1 = fminf(fmaxf(x1, 0.f), 1023.f);
    y2 = fminf(fmaxf(y2, 0.f), 1023.f);
    x2 = fminf(fmaxf(x2, 0.f), 1023.f);
    boxes[(size_t)i*4 + 0] = y1;
    boxes[(size_t)i*4 + 1] = x1;
    boxes[(size_t)i*4 + 2] = y2;
    boxes[(size_t)i*4 + 3] = x2;
}

// ---------------- NMS phase A: exact top-6000 threshold + gather ----------------
__global__ __launch_bounds__(256) void nms_prep_kernel(
    const float* __restrict__ prob, const float* __restrict__ boxes,
    float* __restrict__ cand)        // [8][6][6000]: y1,x1,y2,x2,area,score
{
    const int b   = blockIdx.x;
    const int tid = threadIdx.x;
    const float* pr = prob  + (size_t)b * NPROP;
    const float* bx = boxes + (size_t)b * NPROP * 4;
    float* C   = cand + (size_t)b * 6 * PRE_NMS;
    float* cy1 = C;               float* cx1 = C + PRE_NMS;
    float* cy2 = C + 2*PRE_NMS;   float* cx2 = C + 3*PRE_NMS;
    float* car = C + 4*PRE_NMS;   float* cs  = C + 5*PRE_NMS;

    __shared__ unsigned red[256];
    __shared__ int scount;

    // binary search on float bit pattern (all probs in (0,1) => monotone as uint)
    unsigned lo = 0u, hi = 0xFFFFFFFFu;
    for (int it = 0; it < 32; ++it) {
        unsigned mid = lo + ((hi - lo) >> 1);
        if (mid == lo) break;
        unsigned cnt = 0;
        for (int k = tid; k < NPROP; k += 256)
            cnt += (__float_as_uint(pr[k]) >= mid) ? 1u : 0u;
        red[tid] = cnt;
        __syncthreads();
        for (int s = 128; s > 0; s >>= 1) {
            if (tid < s) red[tid] += red[tid + s];
            __syncthreads();
        }
        unsigned total = red[0];
        __syncthreads();
        if (total >= (unsigned)PRE_NMS) lo = mid; else hi = mid;
    }

    if (tid == 0) scount = 0;
    __syncthreads();
    const unsigned T = lo;
    for (int k = tid; k < NPROP; k += 256) {
        if (__float_as_uint(pr[k]) >= T) {
            int idx = atomicAdd(&scount, 1);
            if (idx < PRE_NMS) {
                const float y1 = bx[(size_t)k*4+0], x1 = bx[(size_t)k*4+1];
                const float y2 = bx[(size_t)k*4+2], x2 = bx[(size_t)k*4+3];
                cy1[idx] = y1; cx1[idx] = x1; cy2[idx] = y2; cx2[idx] = x2;
                car[idx] = (y2 - y1) * (x2 - x1);
                cs[idx]  = pr[k];
            }
        }
    }
}

// ---------------- NMS phase B: 300 greedy iterations (scores in LDS) ----------------
__global__ __launch_bounds__(256) void nms_iter_kernel(
    const float* __restrict__ cand, float* __restrict__ top_out)  // [8][300][4]
{
    const int b   = blockIdx.x;
    const int tid = threadIdx.x;
    const float* C  = cand + (size_t)b * 6 * PRE_NMS;
    const float* cy1 = C;               const float* cx1 = C + PRE_NMS;
    const float* cy2 = C + 2*PRE_NMS;   const float* cx2 = C + 3*PRE_NMS;
    const float* car = C + 4*PRE_NMS;   const float* cs  = C + 5*PRE_NMS;

    __shared__ float scs[PRE_NMS];
    __shared__ float rv[256];
    __shared__ int   ri[256];
    __shared__ float sbox[5];

    for (int k = tid; k < PRE_NMS; k += 256) scs[k] = cs[k];
    __syncthreads();

    for (int i = 0; i < POST_NMS; ++i) {
        // argmax with first-index tie-break (matches jnp.argmax)
        float best = -__builtin_inff();
        int   bidx = 0x7FFFFFFF;
        for (int k = tid; k < PRE_NMS; k += 256) {
            const float v = scs[k];
            if (v > best) { best = v; bidx = k; }
        }
        rv[tid] = best; ri[tid] = bidx;
        __syncthreads();
        for (int s = 128; s > 0; s >>= 1) {
            if (tid < s) {
                const float v2 = rv[tid + s]; const int i2 = ri[tid + s];
                if (v2 > rv[tid] || (v2 == rv[tid] && i2 < ri[tid])) { rv[tid] = v2; ri[tid] = i2; }
            }
            __syncthreads();
        }
        const int j = ri[0];
        __syncthreads();
        if (tid == 0) {
            const float y1 = cy1[j], x1 = cx1[j], y2 = cy2[j], x2 = cx2[j];
            top_out[((size_t)b*POST_NMS + i)*4 + 0] = floorf(y1);
            top_out[((size_t)b*POST_NMS + i)*4 + 1] = floorf(x1);
            top_out[((size_t)b*POST_NMS + i)*4 + 2] = floorf(y2);
            top_out[((size_t)b*POST_NMS + i)*4 + 3] = floorf(x2);
            sbox[0] = y1; sbox[1] = x1; sbox[2] = y2; sbox[3] = x2; sbox[4] = car[j];
        }
        __syncthreads();
        const float py1 = sbox[0], px1 = sbox[1], py2 = sbox[2], px2 = sbox[3], pa = sbox[4];
        for (int k = tid; k < PRE_NMS; k += 256) {
            const float yy1 = fmaxf(py1, cy1[k]);
            const float xx1 = fmaxf(px1, cx1[k]);
            const float yy2 = fminf(py2, cy2[k]);
            const float xx2 = fminf(px2, cx2[k]);
            const float inter = fmaxf(yy2 - yy1, 0.f) * fmaxf(xx2 - xx1, 0.f);
            const float iou = inter / (pa + car[k] - inter + 1e-9f);
            if (iou >= 0.7f) scs[k] = -__builtin_inff();
        }
        __syncthreads();
    }
}

// ---------------- launcher ----------------
extern "C" void kernel_launch(void* const* d_in, const int* in_sizes, int n_in,
                              void* d_out, int out_size, void* d_ws, size_t ws_size,
                              hipStream_t stream) {
    (void)in_sizes; (void)n_in; (void)out_size; (void)ws_size;
    const float* features = (const float*)d_in[0];   // [8][512][64][64]
    const float* conv_w   = (const float*)d_in[1];   // [512][512][3][3]
    const float* conv_b   = (const float*)d_in[2];   // [512]
    const float* cls_w    = (const float*)d_in[3];   // [18][512]
    const float* cls_b    = (const float*)d_in[4];   // [18]
    const float* bbox_w   = (const float*)d_in[5];   // [36][512]
    const float* bbox_b   = (const float*)d_in[6];   // [36]

    char* ws = (char*)d_ws;
    _Float16* feat_h = (_Float16*)(ws + OFF_FEATH);
    _Float16* wp     = (_Float16*)(ws + OFF_WP);
    _Float16* rpn_h  = (_Float16*)(ws + OFF_RPN);
    _Float16* wp1    = (_Float16*)(ws + OFF_WP1);
    float*    out1   = (float*)   (ws + OFF_OUT1);
    float*    prob   = (float*)   (ws + OFF_PROB);
    float*    boxes  = (float*)   (ws + OFF_BOX);
    float*    cand   = (float*)   (ws + OFF_CAND);

    float* top_out   = (float*)d_out;                       // [8][300][4]
    float* score_out = (float*)d_out + 8*POST_NMS*4;        // [8][36864][2]
    float* tgt_out   = score_out + (size_t)8*NPROP*2;       // [8][36864][4]

    const int nfeat = 8*512*NPOS;
    cvt_feat_kernel<<<(nfeat + 255)/256, 256, 0, stream>>>(features, feat_h, nfeat);
    pack_convw_kernel<<<(32*9*16*32*16 + 255)/256, 256, 0, stream>>>(conv_w, wp);
    pack_w1_kernel<<<(16*4*32*16 + 255)/256, 256, 0, stream>>>(cls_w, bbox_w, wp1);

    conv3x3_kernel<<<dim3(4, 64, 8), 256, 0, stream>>>(feat_h, wp, conv_b, rpn_h);
    conv1x1_kernel<<<dim3(32, 8), 256, 0, stream>>>(rpn_h, wp1, cls_b, bbox_b, out1);

    decode_kernel<<<(8*NPROP + 255)/256, 256, 0, stream>>>(out1, score_out, tgt_out, prob, boxes);

    nms_prep_kernel<<<8, 256, 0, stream>>>(prob, boxes, cand);
    nms_iter_kernel<<<8, 256, 0, stream>>>(cand, top_out);
}
